// MGEmbedding_29411936043440
// MI455X (gfx1250) — matrix operand order repaired and checked by reading, not compiled, not run on hardware
//
#include <hip/hip_runtime.h>

// ---------------------------------------------------------------------------
// FiLM-MLP over gathered embeddings, MI455X (gfx1250), wave32 + WMMA f16.
//   g   = embeddings[patch_idx]            [R,128]   (R = B*P = 393216)
//   h   = gelu(g @ W1 + b1)                [R,128]
//   o   = h @ W2 + b2                      [R,256]
//   out = x * o[:, :128] + o[:, 128:]      [R,128]
// HBM-bound (~0.6 GB traffic ≈ 26 us at 23.3 TB/s); math on
// v_wmma_f32_16x16x32_f16 (96 WMMAs / wave), GELU on native v_tanh_f32.
// ---------------------------------------------------------------------------

typedef __attribute__((ext_vector_type(16))) _Float16 v16h;
typedef __attribute__((ext_vector_type(8)))  _Float16 v8h;
typedef __attribute__((ext_vector_type(4)))  _Float16 v4h;
typedef __attribute__((ext_vector_type(8)))  float    v8f;
typedef __attribute__((ext_vector_type(4)))  float    f4;

#define FDIM 128

__device__ __forceinline__ float gelu_tanh(float v) {
    // jax.nn.gelu (approximate=True): 0.5*v*(1+tanh(sqrt(2/pi)*(v+0.044715 v^3)))
    const float c0 = 0.7978845608028654f;
    const float c1 = 0.044715f;
    float u = c0 * v * (1.0f + c1 * v * v);
#if __has_builtin(__builtin_amdgcn_tanhf)
    float t = __builtin_amdgcn_tanhf(u);      // gfx1250 native v_tanh_f32
#else
    float t = tanhf(u);
#endif
    return 0.5f * v * (1.0f + t);
}

__global__ __launch_bounds__(256) void film_mlp_wmma(
    const float* __restrict__ x,          // [R,128]
    const float* __restrict__ emb,        // [N,128]
    const float* __restrict__ w1,         // [128,128] row-major (k,n)
    const float* __restrict__ b1,         // [128]
    const float* __restrict__ w2,         // [128,256] row-major (k,n)
    const float* __restrict__ b2,         // [256]
    const int*  __restrict__ patch_idx,   // [R]
    float* __restrict__ out)              // [R,128]
{
    // 128 KB LDS total: 2 concurrent WGs per WGP even in CU mode.
    __shared__ _Float16 sGH[8][16][FDIM];      // 32 KB: g tile, reused for H
    __shared__ _Float16 sW1T[FDIM][FDIM];      // 32 KB: W1 transposed [n][k], f16
    __shared__ _Float16 sW2T[2 * FDIM][FDIM];  // 64 KB: W2 transposed [n][k], f16

    const int tid  = threadIdx.x;
    const int wid  = tid >> 5;      // wave 0..7, owns 16 rows
    const int lane = tid & 31;
    const int l16  = lane & 15;
    const int hi   = lane >> 4;     // 0: lanes 0-15, 1: lanes 16-31
    const size_t rowBase = (size_t)blockIdx.x * 128;

    // Prefetch this WG's 64 KB x tile (global_prefetch_b8) so the FiLM-phase
    // loads at the end of the kernel hit cache instead of raw HBM latency.
    {
        const char* xb = (const char*)(x + rowBase * FDIM);
        __builtin_prefetch(xb + tid * 256, 0, 0);
        __builtin_prefetch(xb + tid * 256 + 128, 0, 0);
    }

    // ---- stage W1^T, W2^T into LDS as f16 (once per WG; weights are L2-hot)
    for (int i = tid * 4; i < FDIM * FDIM; i += 256 * 4) {
        f4 v = *(const f4*)(w1 + i);
        int k = i >> 7, n = i & 127;
        sW1T[n + 0][k] = (_Float16)v.x;
        sW1T[n + 1][k] = (_Float16)v.y;
        sW1T[n + 2][k] = (_Float16)v.z;
        sW1T[n + 3][k] = (_Float16)v.w;
    }
    for (int i = tid * 4; i < FDIM * 2 * FDIM; i += 256 * 4) {
        f4 v = *(const f4*)(w2 + i);
        int k = i >> 8, n = i & 255;
        sW2T[n + 0][k] = (_Float16)v.x;
        sW2T[n + 1][k] = (_Float16)v.y;
        sW2T[n + 2][k] = (_Float16)v.z;
        sW2T[n + 3][k] = (_Float16)v.w;
    }

    // ---- gather 128 embedding rows -> LDS f16 (2 threads per 512B row)
    {
        int r  = tid >> 1;              // 0..127
        int c0 = (tid & 1) * 64;
        size_t node = (size_t)patch_idx[rowBase + r];
        const f4* src = (const f4*)(emb + node * (size_t)FDIM + c0);
        _Float16* dst = &sGH[r >> 4][r & 15][c0];
        #pragma unroll
        for (int j = 0; j < 16; ++j) {
            f4 v = src[j];
            v4h h = {(_Float16)v.x, (_Float16)v.y, (_Float16)v.z, (_Float16)v.w};
            *(v4h*)(dst + j * 4) = h;
        }
    }
    __syncthreads();

    // A fragment (16x32 f16): lanes 0-15 = rows, half-pairs K {c..c+7, c+16..c+23},
    // lanes 16-31 shifted by +8 in K (ISA 16-bit A layout).
    auto loadA = [&](const _Float16* rowPtr, int kc) -> v16h {
        int off = kc * 32 + hi * 8;
        v8h lo = *(const v8h*)(rowPtr + off);
        v8h up = *(const v8h*)(rowPtr + off + 16);
        v16h a;
        #pragma unroll
        for (int i = 0; i < 8; ++i) { a[i] = lo[i]; a[i + 8] = up[i]; }
        return a;
    };
    // B fragment (32x16 f16): lane = column n, K halves {c..c+7, c+16..c+23}
    // (lanes 16-31 shifted by +8 in K, matching A).
    auto loadB = [&](const _Float16* wT, int n, int kc) -> v16h {
        const _Float16* p = wT + (size_t)n * FDIM + kc * 32 + hi * 8;
        v8h lo = *(const v8h*)(p);
        v8h up = *(const v8h*)(p + 16);
        v16h b;
        #pragma unroll
        for (int i = 0; i < 8; ++i) { b[i] = lo[i]; b[i + 8] = up[i]; }
        return b;
    };

    const _Float16* myRow = &sGH[wid][l16][0];

    // ---- matmul1: H = gelu(G @ W1 + b1), 16x128 per wave
    v16h aG[4];
    #pragma unroll
    for (int kc = 0; kc < 4; ++kc) aG[kc] = loadA(myRow, kc);

    v8f acc1[8];
    #pragma unroll
    for (int nt = 0; nt < 8; ++nt) {
        v8f c = {};
        #pragma unroll
        for (int kc = 0; kc < 4; ++kc) {
            v16h b = loadB(&sW1T[0][0], nt * 16 + l16, kc);
            c = __builtin_amdgcn_wmma_f32_16x16x32_f16(
                    false, aG[kc], false, b, (short)0, c, false, false);
        }
        acc1[nt] = c;
    }

    // bias + gelu, write H into this wave's own LDS region (g is dead now;
    // wave-private region -> only wave-local DS ordering needed).
    #pragma unroll
    for (int nt = 0; nt < 8; ++nt) {
        int col = nt * 16 + l16;
        float bias = b1[col];
        #pragma unroll
        for (int v = 0; v < 8; ++v) {
            float h = gelu_tanh(acc1[nt][v] + bias);   // C layout: row v + 8*hi
            sGH[wid][v + hi * 8][col] = (_Float16)h;
        }
    }

    // ---- matmul2 + fused FiLM: o = H @ W2 + b2; out = x*scale + shift
    v16h aH[4];
    #pragma unroll
    for (int kc = 0; kc < 4; ++kc) aH[kc] = loadA(myRow, kc);

    #pragma unroll
    for (int nt = 0; nt < 8; ++nt) {
        v8f cs = {};   // scale columns  nt*16 .. nt*16+15
        v8f cf = {};   // shift columns 128+nt*16 ..
        #pragma unroll
        for (int kc = 0; kc < 4; ++kc) {
            v16h bs = loadB(&sW2T[0][0], nt * 16 + l16, kc);
            v16h bf = loadB(&sW2T[0][0], FDIM + nt * 16 + l16, kc);
            cs = __builtin_amdgcn_wmma_f32_16x16x32_f16(
                     false, aH[kc], false, bs, (short)0, cs, false, false);
            cf = __builtin_amdgcn_wmma_f32_16x16x32_f16(
                     false, aH[kc], false, bf, (short)0, cf, false, false);
        }
        int col = nt * 16 + l16;
        float bS = b2[col];
        float bF = b2[FDIM + col];
        #pragma unroll
        for (int v = 0; v < 8; ++v) {
            size_t row = rowBase + (size_t)wid * 16 + v + hi * 8;
            size_t idx = row * FDIM + col;          // 64B-coalesced per 16 lanes
            out[idx] = x[idx] * (cs[v] + bS) + (cf[v] + bF);
        }
    }
}

extern "C" void kernel_launch(void* const* d_in, const int* in_sizes, int n_in,
                              void* d_out, int out_size, void* d_ws, size_t ws_size,
                              hipStream_t stream) {
    // setup_inputs order: x, embeddings, w1, b1, w2, b2, group_idx, patch_idx
    const float* x    = (const float*)d_in[0];
    const float* emb  = (const float*)d_in[1];
    const float* w1   = (const float*)d_in[2];
    const float* b1   = (const float*)d_in[3];
    const float* w2   = (const float*)d_in[4];
    const float* b2   = (const float*)d_in[5];
    const int*   pidx = (const int*)d_in[7];     // flattened [B,P] == row order
    float* out = (float*)d_out;

    int R = in_sizes[7];          // B*P = 393216, divisible by 128
    int nBlocks = R / 128;        // 3072 workgroups of 256 threads (8 waves)
    film_mlp_wmma<<<nBlocks, 256, 0, stream>>>(x, emb, w1, b1, w2, b2, pidx, out);
}
